// CrossRQVAE_13932873908339
// MI455X (gfx1250) — compile-verified, matrix operand order checked by reading, not run on hardware
//
#include <hip/hip_runtime.h>
#include <stdint.h>
#include <stddef.h>

typedef __attribute__((ext_vector_type(16))) __bf16 v16bf;
typedef __attribute__((ext_vector_type(8)))  float  v8f;
typedef __attribute__((ext_vector_type(2)))  float  v2f;
typedef int async_v4i __attribute__((vector_size(16)));   // matches builtin param type

#if defined(__has_builtin)
#  if __has_builtin(__builtin_amdgcn_global_load_async_to_lds_b128) && \
      __has_builtin(__builtin_amdgcn_s_wait_asynccnt)
#    define USE_ASYNC 1
#  endif
#endif
#ifndef USE_ASYNC
#  define USE_ASYNC 0
#endif

namespace {

constexpr int BROWS = 32768;   // batch
constexpr int EDIM  = 32;      // embedding dim
constexpr int NCODE = 256;     // codes per codebook
constexpr int NQ    = 3;       // quantizers

// ---- output layout (floats) ----
constexpr size_t OFF_OUT    = 0;
constexpr size_t OFF_LOSS   = (size_t)BROWS * 1024;                 // 33554432
constexpr size_t OFF_IDX    = OFF_LOSS + 1;                         // 33554433
constexpr size_t OFF_OH     = OFF_IDX + (size_t)BROWS * NQ;         // 33652737
constexpr size_t OFF_LOGITS = OFF_OH + (size_t)BROWS * NQ * NCODE;  // 58818561

__device__ __forceinline__ uint32_t bf16bits(float f) {
  uint32_t u = __float_as_uint(f);
  return (u + 0x7FFFu + ((u >> 16) & 1u)) >> 16;   // RNE f32->bf16
}
__device__ __forceinline__ uint32_t packbf2(float lo, float hi) {
  return bf16bits(lo) | (bf16bits(hi) << 16);
}

union FragBF { v16bf v; uint4 q[2]; };

// 16-byte global->LDS copy: async DMA on CDNA5, register bounce otherwise.
__device__ __forceinline__ void cp16_g2l(const uint16_t* g, uint16_t* l) {
#if USE_ASYNC
  __builtin_amdgcn_global_load_async_to_lds_b128((async_v4i*)g, (async_v4i*)l, 0, 0);
#else
  *(uint4*)l = *(const uint4*)g;
#endif
}

__device__ __forceinline__ void wait_async_hi() {   // leave <=3 outstanding
#if USE_ASYNC
  __builtin_amdgcn_s_wait_asynccnt(3);
#endif
}
__device__ __forceinline__ void wait_async_hi1() {  // leave <=1 outstanding
#if USE_ASYNC
  __builtin_amdgcn_s_wait_asynccnt(1);
#endif
}
__device__ __forceinline__ void wait_async_zero() {
#if USE_ASYNC
  __builtin_amdgcn_s_wait_asynccnt(0);
#endif
}

// ============================================================================
// WMMA GEMM: Y[M,N] = act(X[M,K] * W[N,K]^T + bias),  M = BROWS
// W is bf16 (pre-converted). X is bf16 (XBF) or f32. Y is bf16 (YBF) or f32.
// 256 threads / 8 waves, tile 128 x BN, double-buffered LDS, async staging.
// ============================================================================
template <int K, int N, bool RELU, bool XBF, bool YBF>
__launch_bounds__(256, 2)
__global__ void gemm_bias_act(const void* __restrict__ Xv,
                              const uint16_t* __restrict__ W,
                              const float* __restrict__ bias,
                              void* __restrict__ Yv) {
  constexpr int BM  = 128;
  constexpr int BN  = (N >= 64) ? 64 : 32;
  constexpr int KT  = 32;
  constexpr int SA  = 48;          // LDS row stride (bf16) -> 96B rows, 16B aligned
  constexpr int WGN = BN / 32;
  constexpr int WGM = 8 / WGN;
  constexpr int WTM = BM / WGM;
  constexpr int WM  = WTM / 16;
  constexpr int NTILES = K / KT;
  constexpr int BCPW = (BN * 4) / 8;     // B-tile 16B chunks per wave

  __shared__ __align__(16) uint16_t As[2][BM * SA];
  __shared__ __align__(16) uint16_t Bs[2][BN * SA];

  const int tid  = threadIdx.x;
  const int lane = tid & 31;
  const int wave = tid >> 5;
  const int wgm  = wave % WGM;
  const int wgn  = wave / WGM;
  const int m0   = blockIdx.x * BM;
  const int n0   = blockIdx.y * BN;

  const uint16_t* Xb = (const uint16_t*)Xv;
  const float*    Xf = (const float*)Xv;

  // issue async copies for one K-tile into LDS buffer `buf`
  auto issue = [&](int buf, int k0) {
    {  // B tile: exactly one async op per wave (duplicated lanes when BN==32)
      const int i = wave * BCPW + (lane % BCPW);
      const int r = i >> 2, c = i & 3;
      cp16_g2l(W + (size_t)(n0 + r) * K + k0 + c * 8, &Bs[buf][r * SA + c * 8]);
    }
    if (XBF) {  // A tile: two async ops per wave
#pragma unroll
      for (int j = 0; j < 2; ++j) {
        const int i = wave * 64 + j * 32 + lane;
        const int r = i >> 2, c = i & 3;
        cp16_g2l(Xb + (size_t)(m0 + r) * K + k0 + c * 8, &As[buf][r * SA + c * 8]);
      }
    }
  };
  // f32 A staging (enc0 / dec0): load, convert to bf16, store to LDS
  auto stageA_f32 = [&](int buf, int k0) {
#pragma unroll
    for (int i = tid; i < BM * 8; i += 256) {
      const int r = i >> 3;
      const int c = (i & 7) * 4;
      const float4 f = *(const float4*)(Xf + (size_t)(m0 + r) * K + k0 + c);
      uint2 p; p.x = packbf2(f.x, f.y); p.y = packbf2(f.z, f.w);
      *(uint2*)(&As[buf][r * SA + c]) = p;
    }
  };

  v8f acc[WM][2];
#pragma unroll
  for (int a = 0; a < WM; ++a) {
#pragma unroll
    for (int b = 0; b < 2; ++b) acc[a][b] = {};
  }

  const int fr = lane & 15;
  const int fh = lane >> 4;

  issue(0, 0);
  if (!XBF) stageA_f32(0, 0);

  for (int t = 0; t < NTILES; ++t) {
    const int cur = t & 1;
    if (t + 1 < NTILES) {
      issue(cur ^ 1, (t + 1) * KT);
      if (!XBF) stageA_f32(cur ^ 1, (t + 1) * KT);
      if (XBF) wait_async_hi(); else wait_async_hi1();   // tile t's copies done
    } else {
      wait_async_zero();
    }
    __syncthreads();

    FragBF bf[2];
#pragma unroll
    for (int tn = 0; tn < 2; ++tn) {
      const int col = wgn * 32 + tn * 16 + fr;
      bf[tn].q[0] = *(const uint4*)(&Bs[cur][col * SA + 8 * fh]);
      bf[tn].q[1] = *(const uint4*)(&Bs[cur][col * SA + 16 + 8 * fh]);
    }
#pragma unroll
    for (int tm = 0; tm < WM; ++tm) {
      const int row = wgm * WTM + tm * 16 + fr;
      FragBF af;
      af.q[0] = *(const uint4*)(&As[cur][row * SA + 8 * fh]);
      af.q[1] = *(const uint4*)(&As[cur][row * SA + 16 + 8 * fh]);
#pragma unroll
      for (int tn = 0; tn < 2; ++tn) {
        acc[tm][tn] = __builtin_amdgcn_wmma_f32_16x16x32_bf16(
            false, af.v, false, bf[tn].v, (short)0, acc[tm][tn], false, false);
      }
    }
    __syncthreads();
  }

  // store: C layout — lane gives N, VGPR v gives M = v + 8*(lane>=16)
  const int mo = fh * 8;
  float*    Yf = (float*)Yv;
  uint16_t* Yb = (uint16_t*)Yv;
#pragma unroll
  for (int tm = 0; tm < WM; ++tm) {
#pragma unroll
    for (int tn = 0; tn < 2; ++tn) {
      const int gn   = n0 + wgn * 32 + tn * 16 + fr;
      const float bv = bias[gn];
      const int gmb  = m0 + wgm * WTM + tm * 16 + mo;
#pragma unroll
      for (int v = 0; v < 8; ++v) {
        float val = acc[tm][tn][v] + bv;
        if (RELU) val = fmaxf(val, 0.0f);
        const size_t gi = (size_t)(gmb + v) * N + gn;
        if (YBF) Yb[gi] = (uint16_t)bf16bits(val);
        else     Yf[gi] = val;
      }
    }
  }
}

// ============================================================================
// f32 -> bf16 bulk conversion (weights prepass); n multiple of 4
// ============================================================================
__global__ void f32_to_bf16_kernel(const float* __restrict__ src,
                                   uint16_t* __restrict__ dst, int n) {
  const int i = (blockIdx.x * 256 + threadIdx.x) * 4;
  if (i < n) {
    const float4 f = *(const float4*)(src + i);
    uint2 p; p.x = packbf2(f.x, f.y); p.y = packbf2(f.z, f.w);
    *(uint2*)(dst + i) = p;
  }
}

// ============================================================================
// One residual-VQ step (full f32). Block = 16 rows, codebook staged in LDS.
// d[b,n] = ||r_b||^2 + ||c_n||^2 - 2 r_b.c_n  via V_WMMA_F32_16X16X4_F32
// ============================================================================
__launch_bounds__(256, 1)
__global__ void vq_step_kernel(const float* res_in,
                               const float* __restrict__ cb,
                               float* res_out,
                               float* __restrict__ xq_acc,
                               float* __restrict__ logits_out,
                               float* __restrict__ oh_out,
                               int*   __restrict__ idx_out,
                               float* __restrict__ loss_acc,
                               int q, int first) {
  constexpr int SC = 36;
  constexpr int SD = 260;

  __shared__ __align__(16) float cbs[NCODE * SC];
  __shared__ float c2[NCODE];
  __shared__ __align__(16) float res[16 * EDIM];
  __shared__ float r2[16];
  __shared__ float dist[16 * SD];
  __shared__ int   idxs[16];
  __shared__ float red[256];

  const int tid  = threadIdx.x;
  const int lane = tid & 31;
  const int wave = tid >> 5;
  const int m0   = blockIdx.x * 16;

  {
    float s = 0.f;
#pragma unroll
    for (int j = 0; j < 8; ++j) {
      const float4 f = *(const float4*)(cb + (size_t)tid * EDIM + j * 4);
      cbs[tid * SC + j * 4 + 0] = f.x;
      cbs[tid * SC + j * 4 + 1] = f.y;
      cbs[tid * SC + j * 4 + 2] = f.z;
      cbs[tid * SC + j * 4 + 3] = f.w;
      s += f.x * f.x + f.y * f.y + f.z * f.z + f.w * f.w;
    }
    c2[tid] = s;
  }
  for (int i = tid; i < 16 * EDIM; i += 256)
    res[i] = res_in[(size_t)m0 * EDIM + i];
  __syncthreads();
  if (tid < 16) {
    float s = 0.f;
    for (int k = 0; k < EDIM; ++k) { const float v = res[tid * EDIM + k]; s += v * v; }
    r2[tid] = s;
  }
  __syncthreads();

  const int fr = lane & 15;
  const int fh = lane >> 4;

  for (int t = 0; t < 2; ++t) {
    const int n0 = (wave * 2 + t) * 16;
    v8f accd = {};
#if __has_builtin(__builtin_amdgcn_wmma_f32_16x16x4_f32)
#pragma unroll
    for (int kk = 0; kk < 8; ++kk) {
      const int k = kk * 4 + fh * 2;
      v2f a, b;
      a.x = res[fr * EDIM + k];       a.y = res[fr * EDIM + k + 1];
      b.x = cbs[(n0 + fr) * SC + k];  b.y = cbs[(n0 + fr) * SC + k + 1];
      accd = __builtin_amdgcn_wmma_f32_16x16x4_f32(
          false, a, false, b, (short)0, accd, false, false);
    }
#else
#pragma unroll
    for (int v = 0; v < 8; ++v) {
      const int m = v + (fh ? 8 : 0);
      const int n = n0 + fr;
      float s = 0.f;
      for (int k = 0; k < EDIM; ++k) s += res[m * EDIM + k] * cbs[n * SC + k];
      accd[v] = s;
    }
#endif
    const int n = n0 + fr;
#pragma unroll
    for (int v = 0; v < 8; ++v) {
      const int m = v + (fh ? 8 : 0);
      dist[m * SD + n] = r2[m] + c2[n] - 2.0f * accd[v];
    }
  }
  __syncthreads();

  for (int i = tid; i < 16 * NCODE; i += 256) {
    const int m = i >> 8, n = i & 255;
    logits_out[((size_t)(m0 + m) * NQ + q) * NCODE + n] = dist[m * SD + n];
  }
  if (tid < 16) {
    float best = dist[tid * SD];
    int bi = 0;
    for (int n = 1; n < NCODE; ++n) {
      const float d = dist[tid * SD + n];
      if (d < best) { best = d; bi = n; }
    }
    idxs[tid] = bi;
    idx_out[(size_t)(m0 + tid) * NQ + q] = bi;
  }
  __syncthreads();

#pragma unroll
  for (int m = 0; m < 16; ++m)
    oh_out[((size_t)(m0 + m) * NQ + q) * NCODE + tid] = (tid == idxs[m]) ? 1.0f : 0.0f;

  float part = 0.f;
  for (int i = tid; i < 16 * EDIM; i += 256) {
    const int m = i >> 5, k = i & 31;
    const float xq = cbs[idxs[m] * SC + k];
    const float r  = res[m * EDIM + k];
    const float d  = xq - r;
    part += d * d;
    const size_t g = (size_t)(m0 + m) * EDIM + k;
    res_out[g] = r - xq;
    if (first) xq_acc[g] = xq; else xq_acc[g] += xq;
  }
  red[tid] = part;
  __syncthreads();
#pragma unroll
  for (int s = 128; s > 0; s >>= 1) {
    if (tid < s) red[tid] += red[tid + s];
    __syncthreads();
  }
  if (tid == 0) atomicAdd(loss_acc, red[0]);
}

__global__ void zero_loss_kernel(float* p) { if (threadIdx.x == 0) p[0] = 0.0f; }

__global__ void finalize_loss_kernel(const float* __restrict__ loss_acc,
                                     float* __restrict__ out_loss) {
  if (threadIdx.x == 0)
    out_loss[0] = loss_acc[0] * (1.25f / ((float)NQ * (float)BROWS * (float)EDIM));
}

} // namespace

extern "C" void kernel_launch(void* const* d_in, const int* in_sizes, int n_in,
                              void* d_out, int out_size, void* d_ws, size_t ws_size,
                              hipStream_t stream) {
  const float* x      = (const float*)d_in[0];
  const float* enc_w[4] = {(const float*)d_in[1], (const float*)d_in[3],
                           (const float*)d_in[5], (const float*)d_in[7]};
  const float* enc_b[4] = {(const float*)d_in[2], (const float*)d_in[4],
                           (const float*)d_in[6], (const float*)d_in[8]};
  const float* dec_w[4] = {(const float*)d_in[9],  (const float*)d_in[11],
                           (const float*)d_in[13], (const float*)d_in[15]};
  const float* dec_b[4] = {(const float*)d_in[10], (const float*)d_in[12],
                           (const float*)d_in[14], (const float*)d_in[16]};
  const float* cb[NQ] = {(const float*)d_in[17], (const float*)d_in[18], (const float*)d_in[19]};

  float* out = (float*)d_out;
  char*  ws  = (char*)d_ws;
  size_t o = 0;

  // bf16 weight copies
  const int enc_wn[4] = {512 * 1024, 256 * 512, 128 * 256, 32 * 128};
  const int dec_wn[4] = {128 * 32, 256 * 128, 512 * 256, 1024 * 512};
  uint16_t* ew[4]; uint16_t* dw[4];
  for (int i = 0; i < 4; ++i) { ew[i] = (uint16_t*)(ws + o); o += (size_t)enc_wn[i] * 2; }
  for (int i = 0; i < 4; ++i) { dw[i] = (uint16_t*)(ws + o); o += (size_t)dec_wn[i] * 2; }

  uint16_t* bufA = (uint16_t*)(ws + o); o += (size_t)BROWS * 512 * 2;  // enc h1 / dec h3
  uint16_t* bufB = (uint16_t*)(ws + o); o += (size_t)BROWS * 256 * 2;  // enc h2 / dec h2
  uint16_t* bufC = (uint16_t*)(ws + o); o += (size_t)BROWS * 128 * 2;  // enc h3 / dec h1
  float* latent   = (float*)(ws + o); o += (size_t)BROWS * EDIM * 4;
  float* residual = (float*)(ws + o); o += (size_t)BROWS * EDIM * 4;
  float* xq       = (float*)(ws + o); o += (size_t)BROWS * EDIM * 4;
  float* loss_acc = (float*)(ws + o); o += 256;

  const dim3 blk(256);

  // ---- weight f32 -> bf16 prepass (tiny; L2-resident afterwards) ----
  for (int i = 0; i < 4; ++i) {
    f32_to_bf16_kernel<<<dim3((enc_wn[i] / 4 + 255) / 256), blk, 0, stream>>>(enc_w[i], ew[i], enc_wn[i]);
    f32_to_bf16_kernel<<<dim3((dec_wn[i] / 4 + 255) / 256), blk, 0, stream>>>(dec_w[i], dw[i], dec_wn[i]);
  }

  // ---- encoder ----
  gemm_bias_act<1024, 512, true,  false, true ><<<dim3(BROWS / 128, 512 / 64), blk, 0, stream>>>(x,    ew[0], enc_b[0], bufA);
  gemm_bias_act< 512, 256, true,  true,  true ><<<dim3(BROWS / 128, 256 / 64), blk, 0, stream>>>(bufA, ew[1], enc_b[1], bufB);
  gemm_bias_act< 256, 128, true,  true,  true ><<<dim3(BROWS / 128, 128 / 64), blk, 0, stream>>>(bufB, ew[2], enc_b[2], bufC);
  gemm_bias_act< 128,  32, false, true,  false><<<dim3(BROWS / 128, 1),        blk, 0, stream>>>(bufC, ew[3], enc_b[3], latent);

  // ---- residual VQ (f32) ----
  zero_loss_kernel<<<1, 32, 0, stream>>>(loss_acc);
  float* logits_out = out + OFF_LOGITS;
  float* oh_out     = out + OFF_OH;
  int*   idx_out    = (int*)(out + OFF_IDX);
  vq_step_kernel<<<BROWS / 16, blk, 0, stream>>>(latent,   cb[0], residual, xq, logits_out, oh_out, idx_out, loss_acc, 0, 1);
  vq_step_kernel<<<BROWS / 16, blk, 0, stream>>>(residual, cb[1], residual, xq, logits_out, oh_out, idx_out, loss_acc, 1, 0);
  vq_step_kernel<<<BROWS / 16, blk, 0, stream>>>(residual, cb[2], residual, xq, logits_out, oh_out, idx_out, loss_acc, 2, 0);
  finalize_loss_kernel<<<1, 32, 0, stream>>>(loss_acc, out + OFF_LOSS);

  // ---- decoder ----
  gemm_bias_act<  32,  128, true,  false, true ><<<dim3(BROWS / 128, 128 / 64),  blk, 0, stream>>>(xq,   dw[0], dec_b[0], bufC);
  gemm_bias_act< 128,  256, true,  true,  true ><<<dim3(BROWS / 128, 256 / 64),  blk, 0, stream>>>(bufC, dw[1], dec_b[1], bufB);
  gemm_bias_act< 256,  512, true,  true,  true ><<<dim3(BROWS / 128, 512 / 64),  blk, 0, stream>>>(bufB, dw[2], dec_b[2], bufA);
  gemm_bias_act< 512, 1024, false, true,  false><<<dim3(BROWS / 128, 1024 / 64), blk, 0, stream>>>(bufA, dw[3], dec_b[3], out + OFF_OUT);
}
